// VQ_Payam_1451698946501
// MI455X (gfx1250) — matrix operand order repaired and checked by reading, not compiled, not run on hardware
//
#include <hip/hip_runtime.h>
#include <hip/hip_bf16.h>
#include <math.h>

// ---------------------------------------------------------------------------
// VQ-VAE quantization for MI455X (gfx1250, wave32).
// Distances via V_WMMA_F32_16X16X32_BF16 (bf16 A/B, f32 accum):
//   argmin_k ( ||e_k||^2 - 2 * x.e_k )      (row norm dropped; constant per row)
// Codebook staged to LDS with GLOBAL_LOAD_ASYNC_TO_LDS_B128 (ASYNCcnt),
// double-buffered chunks; B tiles software-pipelined one step ahead so
// ds_load latency hides under the WMMA stream.
// ---------------------------------------------------------------------------

#define Bsz 64
#define Tsz 1024
#define Dsz 256
#define Ksz 1024
#define Nrows (Bsz * Tsz)            // 65536
#define NDtot ((size_t)Nrows * Dsz)  // 16777216

// LDS staging: chunk = 128 codes (8 column-tiles) = 64 tiles of 16x32 bf16.
// Tile-major layout: slot [(ctl*8+kt)*32 + lane] is one 32B v16bf block.
#define CT_PER_CHUNK 8
#define CODES_PER_CHUNK 128
#define NCHUNKS (Ksz / CODES_PER_CHUNK)         // 8
#define CHUNK_BYTES (CODES_PER_CHUNK * Dsz * 2) // 65536
#define TILES_PER_CHUNK (CT_PER_CHUNK * 8)      // 64

typedef __attribute__((ext_vector_type(16))) __bf16 v16bf;
typedef __attribute__((ext_vector_type(8)))  float  v8f;

// ------------------------- small utility kernels ---------------------------

__global__ void zero_ws_kernel(unsigned int* __restrict__ counts,
                               float* __restrict__ sumsq) {
    int t = blockIdx.x * blockDim.x + threadIdx.x;
    if (t < Ksz) counts[t] = 0u;
    if (t == 0)  sumsq[0] = 0.0f;
}

// One block (256 threads) per embedding row: bf16 copy + ||e||^2 (exact fp32).
__global__ void prep_embed_kernel(const float* __restrict__ emb,
                                  __bf16* __restrict__ ebf,
                                  float* __restrict__ e2) {
    int k = blockIdx.x;
    int d = threadIdx.x;                       // 0..255
    float v = emb[(size_t)k * Dsz + d];
    ebf[(size_t)k * Dsz + d] = (__bf16)v;
    float s = v * v;
    #pragma unroll
    for (int off = 16; off >= 1; off >>= 1) s += __shfl_xor(s, off);
    __shared__ float ls[8];
    int lane = threadIdx.x & 31, w = threadIdx.x >> 5;
    if (lane == 0) ls[w] = s;
    __syncthreads();
    if (threadIdx.x == 0) {
        float t = 0.f;
        #pragma unroll
        for (int i = 0; i < 8; ++i) t += ls[i];
        e2[k] = t;
    }
}

// ---- async fill of one codebook chunk into LDS (tile-major swizzle) -------
// 4096 16B segments per chunk; 256 threads x 16 async b128 each.
// Segment v -> tile = v>>6, laneSlot = (v&63)>>1, half = v&1.
__device__ __forceinline__ void fill_chunk_async(const char* gebf,
                                                 unsigned ldsBase,
                                                 int cbase, int tid) {
    #pragma unroll
    for (int i = 0; i < 16; ++i) {
        int v      = tid + (i << 8);        // 0..4095
        int tile   = v >> 6;                // ctl*8 + kt
        int within = v & 63;
        int l      = within >> 1;           // destination lane slot 0..31
        int half   = within & 1;
        int ctl    = tile >> 3;
        int kt     = tile & 7;
        int code   = cbase + ctl * 16 + (l & 15);
        int gb     = code * (Dsz * 2) + (l >> 4) * 16 + kt * 64 + half * 32;
        unsigned la = ldsBase + ((unsigned)v << 4);
        unsigned long long ga = (unsigned long long)(gebf + gb);
        asm volatile("global_load_async_to_lds_b128 %0, %1, off"
                     :: "v"(la), "v"(ga) : "memory");
    }
}

// ------------------------------ main argmin --------------------------------
// One wave owns a 16-row strip of X (16x256 bf16 A strip resident in VGPRs),
// block of 8 waves shares LDS-staged codebook chunks; 8 WMMAs per 16-code tile.

__global__ void __launch_bounds__(256)
vq_argmin_kernel(const float* __restrict__ x,
                 const __bf16* __restrict__ ebf,
                 const float* __restrict__ e2,
                 int* __restrict__ idxOut,
                 unsigned int* __restrict__ counts) {
    __shared__ __attribute__((aligned(128))) char lds_buf[2 * CHUNK_BYTES];

    const int tid   = threadIdx.x;
    const int lane  = tid & 31;
    const int wave  = tid >> 5;
    const int rowBase = (blockIdx.x * 8 + wave) * 16;
    const int nlane = lane & 15;      // column within 16-wide tile / A row
    const int hi    = lane >> 4;      // which half of the wave
    const int koff  = hi * 8;         // A-matrix K sub-offset per ISA layout

    const unsigned ldsBase0 = (unsigned)(size_t)&lds_buf[0];

    // ---- kick off async fill of chunk 0 while we convert the A strip ------
    fill_chunk_async((const char*)ebf, ldsBase0, 0, tid);

    // ---- load + convert A strip: rows rowBase..rowBase+15, D=256 ----------
    const float* xr = x + (size_t)(rowBase + nlane) * Dsz;
    v16bf Av[8];
    #pragma unroll
    for (int kt = 0; kt < 8; ++kt) {
        const float* p0 = xr + kt * 32 + koff;       // K = 32kt+koff+{0..7}
        #pragma unroll
        for (int i = 0; i < 8; ++i) {
            Av[kt][i]     = (__bf16)p0[i];           // elements 0..7
            Av[kt][i + 8] = (__bf16)p0[i + 16];      // elements 8..15 (K+16)
        }
    }

    float vmin[8];
    int   vidx[8];
    #pragma unroll
    for (int j = 0; j < 8; ++j) { vmin[j] = 3.4e38f; vidx[j] = 0; }

    for (int c = 0; c < NCHUNKS; ++c) {
        // Drain this wave's outstanding fill of chunk c, then sync block.
        asm volatile("s_wait_asynccnt 0x0" ::: "memory");
        __syncthreads();
        // Safe to refill the other buffer: the barrier above proves every
        // wave finished consuming chunk c-1 (which lived there).
        if (c + 1 < NCHUNKS)
            fill_chunk_async((const char*)ebf,
                             ldsBase0 + (unsigned)(((c + 1) & 1) * CHUNK_BYTES),
                             (c + 1) * CODES_PER_CHUNK, tid);

        const v16bf* base =
            (const v16bf*)&lds_buf[(c & 1) * CHUNK_BYTES] + lane;

        // Flattened 64-tile stream with one-step B prefetch: the ds_loads
        // for tile t+1 issue before wmma(t), so the pre-wmma wait leaves the
        // next pair in flight instead of draining DScnt to zero.
        v8f   acc;
        v16bf Bcur = base[0];
        #pragma unroll
        for (int t = 0; t < TILES_PER_CHUNK; ++t) {
            v16bf Bnext = base[((t + 1) & (TILES_PER_CHUNK - 1)) * 32];
            if ((t & 7) == 0)
                acc = (v8f){0.f, 0.f, 0.f, 0.f, 0.f, 0.f, 0.f, 0.f};
            acc = __builtin_amdgcn_wmma_f32_16x16x32_bf16(
                false, Av[t & 7], false, Bcur, (short)0, acc, false, false);
            if ((t & 7) == 7) {
                const int c0   = c * CODES_PER_CHUNK + (t >> 3) * 16;
                const float e2n  = e2[c0 + nlane];
                const int   cand = c0 + nlane;
                #pragma unroll
                for (int j = 0; j < 8; ++j) {
                    float drel = fmaf(-2.0f, acc[j], e2n); // ||e||^2 - 2 x.e
                    if (drel < vmin[j]) { vmin[j] = drel; vidx[j] = cand; }
                }
            }
            Bcur = Bnext;
        }
        __syncthreads();   // all waves done with buffer (c&1) before refill
    }

    // ---- reduce across the 16 lanes holding each row (xor stays in half) --
    #pragma unroll
    for (int off = 8; off >= 1; off >>= 1) {
        #pragma unroll
        for (int j = 0; j < 8; ++j) {
            float om = __shfl_xor(vmin[j], off);
            int   oi = __shfl_xor(vidx[j], off);
            if (om < vmin[j] || (om == vmin[j] && oi < vidx[j])) {
                vmin[j] = om; vidx[j] = oi;
            }
        }
    }

    // ---- 16 writer lanes emit the 16 row argmins + histogram --------------
    if (nlane < 8) {
        int sel = vidx[0];
        #pragma unroll
        for (int j = 1; j < 8; ++j) if (nlane == j) sel = vidx[j];
        int m = hi * 8 + nlane;                      // row within strip
        idxOut[rowBase + m] = sel;
        atomicAdd(&counts[sel], 1u);
    }
}

// --------------------- gather quantized + MSE loss -------------------------
// Each thread handles 16 consecutive elements of one row (D=256, runs align).

__global__ void gather_loss_kernel(const float* __restrict__ x,
                                   const float* __restrict__ emb,
                                   const int* __restrict__ idx,
                                   float* __restrict__ outq,
                                   float* __restrict__ sumsq) {
    size_t base = ((size_t)blockIdx.x * blockDim.x + threadIdx.x) * 16;
    float s = 0.f;
    if (base < NDtot) {
        int n = (int)(base >> 8);
        int d = (int)(base & 255);
        const float* ep = emb + (size_t)idx[n] * Dsz + d;
        const float* xp = x + base;
        float* op = outq + base;
        #pragma unroll
        for (int i = 0; i < 16; ++i) {
            float q  = ep[i];
            float df = q - xp[i];
            op[i] = q;                    // quantized_st == quantized forward
            s = fmaf(df, df, s);
        }
    }
    #pragma unroll
    for (int off = 16; off >= 1; off >>= 1) s += __shfl_xor(s, off);
    if ((threadIdx.x & 31) == 0) atomicAdd(sumsq, s);
}

// --------------------------- one-hot encodings -----------------------------
// enc base offset is 8-byte aligned (elements 2+ND), so float2 stores are OK.

__global__ void encodings_kernel(const int* __restrict__ idx,
                                 float* __restrict__ enc) {
    size_t t = (size_t)blockIdx.x * blockDim.x + threadIdx.x; // N*K/8 threads
    int n  = (int)(t >> 7);            // K/8 = 128 chunks per row
    int c0 = (int)(t & 127) << 3;
    int id = idx[n];
    float2* p = (float2*)(enc + (size_t)n * Ksz + c0);
    #pragma unroll
    for (int i = 0; i < 4; ++i) {
        float2 v;
        v.x = (c0 + 2 * i     == id) ? 1.0f : 0.0f;
        v.y = (c0 + 2 * i + 1 == id) ? 1.0f : 0.0f;
        p[i] = v;
    }
}

// ------------------------------ finalize -----------------------------------

__global__ void finalize_kernel(const unsigned int* __restrict__ counts,
                                const float* __restrict__ sumsq,
                                float* __restrict__ out_loss,
                                float* __restrict__ out_ppl) {
    __shared__ float red[1024];
    int k = threadIdx.x;
    float p = (float)counts[k] * (1.0f / (float)Nrows);
    red[k] = p * logf(p + 1e-10f);
    __syncthreads();
    for (int s = 512; s >= 1; s >>= 1) {
        if (k < s) red[k] += red[k + s];
        __syncthreads();
    }
    if (k == 0) {
        // q_latent_loss == e_latent_loss numerically -> loss = 1.25 * MSE
        out_loss[0] = 1.25f * sumsq[0] * (1.0f / (float)NDtot);
        out_ppl[0]  = expf(-red[0]);
    }
}

// ------------------------------ launcher -----------------------------------

extern "C" void kernel_launch(void* const* d_in, const int* in_sizes, int n_in,
                              void* d_out, int out_size, void* d_ws, size_t ws_size,
                              hipStream_t stream) {
    const float* x   = (const float*)d_in[0];   // [64,1024,256] fp32
    const float* emb = (const float*)d_in[1];   // [1024,256]    fp32

    float* out = (float*)d_out;
    float* out_loss = out;                       // [1]
    float* out_q    = out + 1;                   // [N*D]
    float* out_ppl  = out + 1 + NDtot;           // [1]
    float* out_enc  = out + 2 + NDtot;           // [N*K], 8B-aligned base

    char* w = (char*)d_ws;
    __bf16*       ebf    = (__bf16*)(w);                 // 524288 B
    float*        e2     = (float*)(w + 524288);         // 4096 B
    int*          idx    = (int*)(w + 528384);           // 262144 B
    unsigned int* counts = (unsigned int*)(w + 790528);  // 4096 B
    float*        sumsq  = (float*)(w + 794624);         // 4 B

    zero_ws_kernel<<<5, 256, 0, stream>>>(counts, sumsq);
    prep_embed_kernel<<<Ksz, Dsz, 0, stream>>>(emb, ebf, e2);
    vq_argmin_kernel<<<Nrows / 128, 256, 0, stream>>>(x, ebf, e2, idx, counts);
    gather_loss_kernel<<<(unsigned)(NDtot / 16 / 256), 256, 0, stream>>>(
        x, emb, idx, out_q, sumsq);
    encodings_kernel<<<(unsigned)((size_t)Nrows * Ksz / 8 / 256), 256, 0, stream>>>(
        idx, out_enc);
    finalize_kernel<<<1, 1024, 0, stream>>>(counts, sumsq, out_loss, out_ppl);
}
